// SGNHeadOne_59021440581850
// MI455X (gfx1250) — matrix operand order repaired and checked by reading, not compiled, not run on hardware
//
#include <hip/hip_runtime.h>
#include <stdint.h>

// ---------------------------------------------------------------------------
// MI455X (gfx1250) implementation: every GEMM/conv runs through
// v_wmma_f32_16x16x32_bf16 (bf16 in, f32 accumulate). Weights are pre-packed
// into the exact WMMA A-fragment lane layout; activations are staged in LDS
// in K-pair-packed form so B fragments are 8x ds_load_b32.
// ---------------------------------------------------------------------------

typedef __attribute__((ext_vector_type(16))) __bf16    v16bf;
typedef __attribute__((ext_vector_type(8)))  float     v8f;
typedef __attribute__((ext_vector_type(8)))  uint32_t  v8u;

#define NVOX (128 * 128 * 16)   // H*W*Z voxels

__device__ __forceinline__ unsigned short f2bf(float f) {
  union { float f; uint32_t u; } c; c.f = f;
  uint32_t u = c.u;
  uint32_t r = u + 0x7fffu + ((u >> 16) & 1u);   // round-to-nearest-even
  return (unsigned short)(r >> 16);
}

// ---------------------------------------------------------------------------
// Weight pre-pack: produce A fragments in contiguous K order (k = 16h + s, same as B).
//   lane 0-15  (h=0): VGPR j -> K = 2j,2j+1
//   lane 16-31 (h=1): same +16
// apack layout: [MB][nchunk][lane(32)][dword(8)], chunk = tap*(Cin/32)+ci0/32
// wTransposed==0: w[(m*Cin + ci)*taps + t]   (conv weights [Cout][Cin][27])
// wTransposed==1: w[ci*CoutReal + m]         (MLP weights [Cin][Cout], taps=1)
// ---------------------------------------------------------------------------
__global__ __launch_bounds__(256)
void pack_weights(const float* __restrict__ w, uint32_t* __restrict__ apack,
                  int CoutReal, int Cin, int taps, int MB, int wTransposed) {
  int nchunk = taps * (Cin >> 5);
  int total  = MB * nchunk * 256;
  int idx = blockIdx.x * 256 + threadIdx.x;
  if (idx >= total) return;
  int j    = idx & 7;
  int lane = (idx >> 3) & 31;
  int kc   = (idx >> 8) % nchunk;
  int mb   = (idx >> 8) / nchunk;
  int m    = mb * 16 + (lane & 15);
  int h    = lane >> 4;
  int k0   = 2 * j + 16 * h;
  int t    = kc / (Cin >> 5);
  int c0   = (kc % (Cin >> 5)) * 32;
  uint32_t out = 0;
  for (int p = 0; p < 2; ++p) {
    int ci = c0 + k0 + p;
    unsigned short b = 0;
    if (m < CoutReal) {
      float wv = wTransposed ? w[(size_t)ci * CoutReal + m]
                             : w[((size_t)m * Cin + ci) * taps + t];
      b = f2bf(wv);
    }
    out |= ((uint32_t)b) << (16 * p);
  }
  apack[idx] = out;
}

// ---------------------------------------------------------------------------
// Generic implicit-GEMM conv via WMMA.
//   MB:    Cout_pad/16 M-blocks,  NBLK: N-tiles of 16 voxels (NT = 16*NBLK)
//   TAPS:  27 (3x3x3 SAME conv) or 1 (1x1 conv / GEMM)
//   EPI:   0 bottleneck (write x_bf16 and occ_in = x*(1+prob) bf16)
//          1 leaky_relu -> bf16
//          2 raw f32 store (guarded by CoutReal)
//          3 masked select: out = mask ? x_bf16 : bf16(prior)
// Input xin: bf16 [Cin][NVOX] channel-major (voxel = (h<<11)|(w<<4)|z)
// ---------------------------------------------------------------------------
template <int MB, int NBLK, int TAPS, int EPI>
__global__ __launch_bounds__(256)
void conv_wmma(const unsigned short* __restrict__ xin,
               const uint32_t* __restrict__ apack,
               const float* __restrict__ bias,
               int Cin, int CoutReal,
               float* __restrict__ out_f32,
               unsigned short* __restrict__ out_bf16,
               const float* __restrict__ prob,
               unsigned short* __restrict__ out_bf16_b,
               const unsigned char* __restrict__ mask,
               const unsigned short* __restrict__ xbf) {
  constexpr int NT    = NBLK * 16;
  constexpr int TILES = MB * NBLK;
  constexpr int TPW   = TILES / 8;           // wave-tiles per wave (>=1)
  __shared__ uint32_t Blds[16 * NT];         // K-pair packed: [kp][n]
  unsigned short* BldsH = (unsigned short*)Blds;

  const int tid   = threadIdx.x;
  const int wave  = tid >> 5;
  const int lane  = tid & 31;
  const int vbase = blockIdx.x * NT;
  const int mb    = wave % MB;
  const int nb0   = (wave / MB) * TPW;
  const int h16   = lane >> 4;
  const int nl    = lane & 15;
  const int nchunk = TAPS * (Cin >> 5);

  v8f acc[TPW];
  v8f zero = {0.f, 0.f, 0.f, 0.f, 0.f, 0.f, 0.f, 0.f};
#pragma unroll
  for (int i = 0; i < TPW; ++i) acc[i] = zero;

  for (int t = 0; t < TAPS; ++t) {
    int dh = 0, dw = 0, dz = 0;
    if (TAPS != 1) { dh = t / 9 - 1; dw = (t / 3) % 3 - 1; dz = t % 3 - 1; }
    for (int c0 = 0; c0 < Cin; c0 += 32) {
      __syncthreads();
      // stage B tile (32 K x NT voxels) into LDS, pair-packed on K
      for (int e = tid; e < 32 * NT; e += 256) {
        int kk = e / NT;
        int n  = e % NT;
        int v  = vbase + n;
        int z  = (v & 15) + dz;
        int ww = ((v >> 4) & 127) + dw;
        int hh = (v >> 11) + dh;
        unsigned short val = 0;
        if ((unsigned)z < 16u && (unsigned)ww < 128u && (unsigned)hh < 128u)
          val = xin[(size_t)(c0 + kk) * NVOX + ((size_t)hh << 11) + (ww << 4) + z];
        BldsH[(kk >> 1) * (2 * NT) + n * 2 + (kk & 1)] = val;
      }
      __syncthreads();
      // A fragment: straight from pre-packed global (2x global_load_b128)
      int kc = t * (Cin >> 5) + (c0 >> 5);
      v8u au = *(const v8u*)(apack + ((size_t)(mb * nchunk + kc) * 32 + lane) * 8);
      v16bf av = __builtin_bit_cast(v16bf, au);
#pragma unroll
      for (int i = 0; i < TPW; ++i) {
        int nb = nb0 + i;
        const uint32_t* bp = Blds + h16 * (8 * NT) + nb * 16 + nl;
        v8u bu;
#pragma unroll
        for (int j = 0; j < 8; ++j) bu[j] = bp[j * NT];   // 8x ds_load_b32
        v16bf bv = __builtin_bit_cast(v16bf, bu);
        acc[i] = __builtin_amdgcn_wmma_f32_16x16x32_bf16(
            false, av, false, bv, (short)0, acc[i], false, false);
      }
    }
  }

  // Epilogue. C/D layout: VGPR r, lanes 0-15 -> M=r, lanes 16-31 -> M=r+8.
#pragma unroll
  for (int i = 0; i < TPW; ++i) {
    int nb  = nb0 + i;
    int vox = vbase + nb * 16 + nl;
#pragma unroll
    for (int r = 0; r < 8; ++r) {
      int co = mb * 16 + r + h16 * 8;
      float val = acc[i][r] + ((co < CoutReal) ? bias[co] : 0.f);
      if (EPI == 0) {
        size_t idx = (size_t)co * NVOX + vox;
        out_bf16[idx]   = f2bf(val);                       // x (bf16 seed feats)
        float p         = prob[vox];
        out_bf16_b[idx] = f2bf(val * (1.f + p));           // occ head input
      } else if (EPI == 1) {
        float v2 = val > 0.f ? val : 0.01f * val;          // leaky_relu
        out_bf16[(size_t)co * NVOX + vox] = f2bf(v2);
      } else if (EPI == 2) {
        if (co < CoutReal) out_f32[(size_t)co * NVOX + vox] = val;
      } else if (EPI == 3) {
        size_t idx = (size_t)co * NVOX + vox;
        unsigned short xv = xbf[idx];
        out_bf16[idx] = mask[vox] ? xv : f2bf(val);        // where(occ, x, prior)
      }
    }
  }
}

// ---------------------------------------------------------------------------
// Small elementwise kernels
// ---------------------------------------------------------------------------
__global__ __launch_bounds__(256)
void f32_to_bf16_kernel(const float* __restrict__ in,
                        unsigned short* __restrict__ out, size_t n) {
  size_t i = (size_t)blockIdx.x * 256 + threadIdx.x;
  if (i < n) out[i] = f2bf(in[i]);
}

__global__ __launch_bounds__(256)
void occ_mask_kernel(const float* __restrict__ logits,
                     unsigned char* __restrict__ mask, int n) {
  int v = blockIdx.x * 256 + threadIdx.x;
  if (v >= n) return;
  // softmax(axis=1)[:,1] > 0.5  <=>  logit1 > logit0
  mask[v] = (logits[(size_t)NVOX + v] > logits[v]) ? 1 : 0;
}

// LayerNorm over the 64-feature axis + leaky_relu -> bf16.
// Both loops FULLY unrolled so vals[] lives in VGPRs (no scratch spill).
__global__ __launch_bounds__(256)
void ln_leaky_kernel(const float* __restrict__ h, const float* __restrict__ g,
                     const float* __restrict__ b,
                     unsigned short* __restrict__ out, int n) {
  int v = blockIdx.x * 256 + threadIdx.x;
  if (v >= n) return;
  float vals[64];
  float s = 0.f, s2 = 0.f;
#pragma unroll
  for (int c = 0; c < 64; ++c) {
    float x = h[(size_t)c * NVOX + v];
    vals[c] = x; s += x; s2 += x * x;
  }
  float mu  = s * (1.f / 64.f);
  float var = s2 * (1.f / 64.f) - mu * mu;
  float inv = rsqrtf(var + 1e-5f);
#pragma unroll
  for (int c = 0; c < 64; ++c) {
    float x = (vals[c] - mu) * inv * g[c] + b[c];
    x = x > 0.f ? x : 0.01f * x;
    out[(size_t)c * NVOX + v] = f2bf(x);
  }
}

// ---------------------------------------------------------------------------
// Launch
// ---------------------------------------------------------------------------
extern "C" void kernel_launch(void* const* d_in, const int* in_sizes, int n_in,
                              void* d_out, int out_size, void* d_ws, size_t ws_size,
                              hipStream_t stream) {
  const float* x3d    = (const float*)d_in[0];
  const float* prob   = (const float*)d_in[1];
  const float* w_bot  = (const float*)d_in[2];
  const float* b_bot  = (const float*)d_in[3];
  const float* w_occ1 = (const float*)d_in[4];
  const float* b_occ1 = (const float*)d_in[5];
  const float* w_occ2 = (const float*)d_in[6];
  const float* b_occ2 = (const float*)d_in[7];
  const float* w_mlp1 = (const float*)d_in[8];
  const float* b_mlp1 = (const float*)d_in[9];
  const float* ln_g   = (const float*)d_in[10];
  const float* ln_b   = (const float*)d_in[11];
  const float* w_mlp2 = (const float*)d_in[12];
  const float* b_mlp2 = (const float*)d_in[13];
  const float* w_sdb  = (const float*)d_in[14];
  const float* b_sdb  = (const float*)d_in[15];
  const float* w_ssc  = (const float*)d_in[16];
  const float* b_ssc  = (const float*)d_in[17];

  char* ws = (char*)d_ws;
  const size_t R64 = (size_t)128 * NVOX * 2;  // 64 MB
  // lifetime-aliased workspace regions (~240 MB total):
  unsigned short* xin_bf   = (unsigned short*)(ws + 0);          // x3d bf16; later vox_bf16
  unsigned short* vox_bf   = xin_bf;                             // alias (xin dead after bot)
  unsigned short* x_bf     = (unsigned short*)(ws + R64);        // bot output bf16; later d_bf16
  unsigned short* d_bf     = x_bf;                               // alias (x_bf dead after mlp2)
  unsigned short* occin_bf = (unsigned short*)(ws + 2 * R64);    // occ head input
  float*          h_f32    = (float*)(ws + 2 * R64);             // alias (occin dead after occ1)
  unsigned short* occh_bf  = (unsigned short*)(ws + 3 * R64);    // occ hidden (32 MB)
  unsigned short* h_bf     = occh_bf;                            // alias (occh dead after occ2)
  size_t o = 3 * R64 + (size_t)64 * NVOX * 2;
  float*         logits = (float*)(ws + o);  o += (size_t)2 * NVOX * 4;
  unsigned char* maskp  = (unsigned char*)(ws + o); o += (size_t)NVOX;
  uint32_t* ap_bot  = (uint32_t*)(ws + o); o += (size_t)8 * 108 * 256 * 4;
  uint32_t* ap_occ1 = (uint32_t*)(ws + o); o += (size_t)4 * 108 * 256 * 4;
  uint32_t* ap_occ2 = (uint32_t*)(ws + o); o += (size_t)1 * 54 * 256 * 4;
  uint32_t* ap_mlp1 = (uint32_t*)(ws + o); o += (size_t)4 * 4 * 256 * 4;
  uint32_t* ap_mlp2 = (uint32_t*)(ws + o); o += (size_t)8 * 2 * 256 * 4;
  uint32_t* ap_sdb  = (uint32_t*)(ws + o); o += (size_t)4 * 108 * 256 * 4;
  uint32_t* ap_ssc  = (uint32_t*)(ws + o); o += (size_t)2 * 2 * 256 * 4;

  // 1. convert input activations to bf16
  {
    size_t n = (size_t)128 * NVOX;
    f32_to_bf16_kernel<<<(unsigned)((n + 255) / 256), 256, 0, stream>>>(x3d, xin_bf, n);
  }
  // 2. pre-pack all weights into WMMA A-fragment layout
  pack_weights<<<8 * 108, 256, 0, stream>>>(w_bot,  ap_bot,  128, 128, 27, 8, 0);
  pack_weights<<<4 * 108, 256, 0, stream>>>(w_occ1, ap_occ1,  64, 128, 27, 4, 0);
  pack_weights<<<1 * 54,  256, 0, stream>>>(w_occ2, ap_occ2,   2,  64, 27, 1, 0);
  pack_weights<<<4 * 4,   256, 0, stream>>>(w_mlp1, ap_mlp1,  64, 128,  1, 4, 1);
  pack_weights<<<8 * 2,   256, 0, stream>>>(w_mlp2, ap_mlp2, 128,  64,  1, 8, 1);
  pack_weights<<<4 * 108, 256, 0, stream>>>(w_sdb,  ap_sdb,   64, 128, 27, 4, 0);
  pack_weights<<<2 * 2,   256, 0, stream>>>(w_ssc,  ap_ssc,   20,  64,  1, 2, 0);

  const int G64  = NVOX / 64;   // 4096
  const int G128 = NVOX / 128;  // 2048
  const int GV   = NVOX / 256;  // 1024

  // 3. bottleneck conv 128->128 (3^3); fuse x_bf16 + occ_in = x*(1+prob)
  conv_wmma<8, 4, 27, 0><<<G64, 256, 0, stream>>>(
      xin_bf, ap_bot, b_bot, 128, 128, nullptr, x_bf, prob, occin_bf, nullptr, nullptr);
  // 4. occ hidden conv 128->64 (3^3) + leaky
  conv_wmma<4, 4, 27, 1><<<G64, 256, 0, stream>>>(
      occin_bf, ap_occ1, b_occ1, 128, 64, nullptr, occh_bf, nullptr, nullptr, nullptr, nullptr);
  // 5. occ logits conv 64->2 (3^3), M padded to 16
  conv_wmma<1, 8, 27, 2><<<G128, 256, 0, stream>>>(
      occh_bf, ap_occ2, b_occ2, 64, 2, logits, nullptr, nullptr, nullptr, nullptr, nullptr);
  // 6. occupancy mask
  occ_mask_kernel<<<GV, 256, 0, stream>>>(logits, maskp, NVOX);
  // 7. MLP GEMM1 (as 1x1 conv 128->64), raw f32 for LayerNorm
  conv_wmma<4, 4, 1, 2><<<G64, 256, 0, stream>>>(
      x_bf, ap_mlp1, b_mlp1, 128, 64, h_f32, nullptr, nullptr, nullptr, nullptr, nullptr);
  // 8. LayerNorm over 64 features + leaky -> bf16
  ln_leaky_kernel<<<GV, 256, 0, stream>>>(h_f32, ln_g, ln_b, h_bf, NVOX);
  // 9. MLP GEMM2 (1x1 conv 64->128) fused with where(mask, x, prior) -> vox
  conv_wmma<8, 4, 1, 3><<<G64, 256, 0, stream>>>(
      h_bf, ap_mlp2, b_mlp2, 64, 128, nullptr, vox_bf, nullptr, nullptr, maskp, x_bf);
  // 10. SDB diffusion conv 128->64 (3^3) + leaky
  conv_wmma<4, 4, 27, 1><<<G64, 256, 0, stream>>>(
      vox_bf, ap_sdb, b_sdb, 128, 64, nullptr, d_bf, nullptr, nullptr, nullptr, nullptr);
  // 11. SSC head 1x1 conv 64->20 (M padded to 32) -> d_out f32
  conv_wmma<2, 4, 1, 2><<<G64, 256, 0, stream>>>(
      d_bf, ap_ssc, b_ssc, 64, 20, (float*)d_out, nullptr, nullptr, nullptr, nullptr, nullptr);
}
